// SelfAttention_26018911879785
// MI455X (gfx1250) — compile-verified, matrix operand order checked by reading, not run on hardware
//
#include <hip/hip_runtime.h>
#include <hip/hip_bf16.h>

// ---------------------------------------------------------------------------
// Causal multi-head self-attention for MI455X (gfx1250), bf16 WMMA pipeline.
//   B=2, T=2048, C=1024, H=16, D=64
// Stages:
//   1) x -> bf16 once; weights fp32[K,N] -> bf16[N,K] (transposed)
//   2) GEMM (WMMA bf16, fp32 acc, async-to-LDS double-buffered):
//        Q,K -> bf16 [B,H,T,D]; V -> bf16 [B,H,D,T]
//   3) flash attention (online softmax, never materializes T x T) -> bf16
//   4) output projection GEMM -> fp32 d_out
// ---------------------------------------------------------------------------

typedef __attribute__((ext_vector_type(16))) __bf16 v16bf;
typedef __attribute__((ext_vector_type(8)))  __bf16 v8bf;
typedef __attribute__((ext_vector_type(8)))  float  v8f;
typedef __attribute__((ext_vector_type(4)))  float  v4f;
typedef __attribute__((ext_vector_type(4)))  int    v4i;

#define B_  2
#define T_  2048
#define C_  1024
#define H_  16
#define D_  64
#define M_  (B_*T_)   // 4096 rows

// gfx1250 async global->LDS path (ASYNCcnt), with synchronous fallback.
#if defined(__HIP_DEVICE_COMPILE__) && defined(__has_builtin)
#if __has_builtin(__builtin_amdgcn_global_load_async_to_lds_b128) && \
    __has_builtin(__builtin_amdgcn_s_wait_asynccnt)
#define USE_ASYNC_LDS 1
#endif
#endif

#if defined(USE_ASYNC_LDS)
// Builtin signature: (v4i AS1* src, v4i AS3* dst, imm offset, imm cpol)
typedef __attribute__((address_space(1))) v4i as1_v4i;
typedef __attribute__((address_space(3))) v4i as3_v4i;
#endif

// Copy 16 bytes global -> LDS (async on gfx1250; sync fallback otherwise).
__device__ __forceinline__ void copy_b128_to_lds(const __bf16* g, __bf16* l) {
#if defined(USE_ASYNC_LDS)
  __builtin_amdgcn_global_load_async_to_lds_b128(
      (as1_v4i*)(uintptr_t)g,
      (as3_v4i*)(uint32_t)(uintptr_t)l,   // low 32 bits = LDS offset
      /*imm offset=*/0, /*cpol=*/0);
#else
  *(v8bf*)l = *(const v8bf*)g;
#endif
}

// D = A*B + C,  A:16x32 bf16, B:32x16 bf16, C/D:16x16 f32
__device__ __forceinline__ v8f wmma_bf16(v16bf a, v16bf b, v8f c) {
  return __builtin_amdgcn_wmma_f32_16x16x32_bf16(
      /*neg_a=*/false, a, /*neg_b=*/false, b,
      /*c_mod=*/(short)0, c, /*reuse_a=*/false, /*reuse_b=*/false);
}

// Assemble a 16-element bf16 fragment from two 16-byte chunks.
__device__ __forceinline__ v16bf mk16(const __bf16* p0, const __bf16* p1) {
  v8bf c0 = *(const v8bf*)p0;
  v8bf c1 = *(const v8bf*)p1;
  v16bf r;
#pragma unroll
  for (int i = 0; i < 8; ++i) { r[i] = c0[i]; r[8 + i] = c1[i]; }
  return r;
}

// ---------------------------------------------------------------------------
// Stage 1a: w fp32 [K=C, N=C] (in,out) -> wt bf16 [N, K]
// ---------------------------------------------------------------------------
__global__ void wt_bf16_kernel(const float* __restrict__ w,
                               __bf16* __restrict__ wt) {
  const int idx = blockIdx.x * 256 + threadIdx.x;   // over N*K = 1M
  const int n = idx >> 10;          // / C_
  const int k = idx & (C_ - 1);
  wt[n * C_ + k] = (__bf16)w[k * C_ + n];
}

// Stage 1b: x fp32 -> bf16 (4 elements / thread)
__global__ void x_bf16_kernel(const float* __restrict__ x,
                              __bf16* __restrict__ xb) {
  const int idx = blockIdx.x * 256 + threadIdx.x;   // over M*C/4
  v4f f = *(const v4f*)(x + (size_t)idx * 4);
  __bf16* p = xb + (size_t)idx * 4;
  p[0] = (__bf16)f[0]; p[1] = (__bf16)f[1];
  p[2] = (__bf16)f[2]; p[3] = (__bf16)f[3];
}

// ---------------------------------------------------------------------------
// Stage 2/4: out = X[M,C]bf16 @ W + bias,  W transposed bf16 [N=C, K=C].
// Block 256 (8 waves), tile 128M x 128N, K-step 32, LDS double-buffered with
// async global->LDS prefetch of the next K-tile overlapped with WMMA.
// Waves: 4 (M) x 2 (N); each wave 2 M-tiles x 4 N-tiles of 16x16.
// mode 0: fp32 [M, C];  mode 1: bf16 [B,H,T,D];  mode 2: bf16 [B,H,D,T]
// ---------------------------------------------------------------------------
__global__ __launch_bounds__(256)
void gemm_kernel(const __bf16* __restrict__ X, const __bf16* __restrict__ Wt,
                 const float* __restrict__ bias, float* __restrict__ outF,
                 __bf16* __restrict__ outB, int mode) {
  __shared__ __align__(16) __bf16 Xs[2][128 * 32];
  __shared__ __align__(16) __bf16 Ws[2][128 * 32];

  const int tid  = threadIdx.x;
  const int wid  = tid >> 5;
  const int lane = tid & 31;
  const int l16  = lane & 15;
  const int hi   = lane >> 4;              // half-wave: K-chunk select
  const int wm   = wid & 3;                // wave M group (0..3) -> 32 rows
  const int wn   = wid >> 2;               // wave N group (0..1) -> 64 cols
  const int rowbase = blockIdx.x * 128;
  const int colbase = blockIdx.y * 128;

  v8f acc[2][4];
#pragma unroll
  for (int i = 0; i < 2; ++i)
#pragma unroll
    for (int j = 0; j < 4; ++j) acc[i][j] = v8f{};

  const int srow = tid >> 1;               // 0..127
  const int skh  = (tid & 1) * 16;         // 0 / 16

  // Issue 4 x b128 copies (2 per operand) for one K-tile into buffer `buf`.
  auto stage = [&](int buf, int k0) {
    const __bf16* xg = X  + (size_t)(rowbase + srow) * C_ + k0 + skh;
    const __bf16* wg = Wt + (size_t)(colbase + srow) * C_ + k0 + skh;
    __bf16* xl = &Xs[buf][srow * 32 + skh];
    __bf16* wl = &Ws[buf][srow * 32 + skh];
    copy_b128_to_lds(xg,     xl);
    copy_b128_to_lds(xg + 8, xl + 8);
    copy_b128_to_lds(wg,     wl);
    copy_b128_to_lds(wg + 8, wl + 8);
  };

  stage(0, 0);                             // prologue: first tile in flight
  for (int k0 = 0; k0 < C_; k0 += 32) {
    const int  buf  = (k0 >> 5) & 1;
    const bool last = (k0 + 32 >= C_);
    if (!last) stage(buf ^ 1, k0 + 32);    // prefetch next tile (other buffer)
#if defined(USE_ASYNC_LDS)
    // Async copies complete in order: <=4 outstanding means the 4 copies of
    // the *current* buffer have landed; the 4 just-issued may still fly.
    if (!last) __builtin_amdgcn_s_wait_asynccnt(4);
    else       __builtin_amdgcn_s_wait_asynccnt(0);
#endif
    __syncthreads();                       // all waves' current tile visible

    v16bf a[2], b[4];
#pragma unroll
    for (int mt = 0; mt < 2; ++mt) {
      const __bf16* p = &Xs[buf][(wm * 32 + mt * 16 + l16) * 32 + hi * 8];
      a[mt] = mk16(p, p + 16);
    }
#pragma unroll
    for (int nt = 0; nt < 4; ++nt) {
      const __bf16* p = &Ws[buf][(wn * 64 + nt * 16 + l16) * 32 + hi * 8];
      b[nt] = mk16(p, p + 16);
    }
#pragma unroll
    for (int mt = 0; mt < 2; ++mt)
#pragma unroll
      for (int nt = 0; nt < 4; ++nt)
        acc[mt][nt] = wmma_bf16(a[mt], b[nt], acc[mt][nt]);
    __syncthreads();                       // done reading `buf` (WAR fence)
  }

  // Epilogue: C/D layout -> memory. lane holds column n, rows r(+8 for hi).
#pragma unroll
  for (int mt = 0; mt < 2; ++mt) {
#pragma unroll
    for (int nt = 0; nt < 4; ++nt) {
      const int n  = colbase + wn * 64 + nt * 16 + l16;
      const float bn = bias[n];
#pragma unroll
      for (int r = 0; r < 8; ++r) {
        const int row = rowbase + wm * 32 + mt * 16 + r + hi * 8;
        const float val = acc[mt][nt][r] + bn;
        if (mode == 0) {
          outF[(size_t)row * C_ + n] = val;
        } else {
          const int bb = row >> 11, t = row & (T_ - 1);
          const int h  = n >> 6,    d = n & (D_ - 1);
          if (mode == 1)
            outB[(((size_t)bb * H_ + h) * T_ + t) * D_ + d] = (__bf16)val;
          else
            outB[(((size_t)bb * H_ + h) * D_ + d) * T_ + t] = (__bf16)val;
        }
      }
    }
  }
}

// ---------------------------------------------------------------------------
// Stage 3: flash attention.
// Grid (T/64, B*H), block 128 (4 waves). Wave w owns q rows [qt*64+w*16, +16).
// Per kv tile of 64: S = Q K^T (8 WMMA), scale+causal mask, online softmax
// (width-16 lane shuffles), P -> LDS bf16 relayout, O += P V (8 WMMA).
// Output written as bf16 (consumed by the bf16 projection GEMM).
// ---------------------------------------------------------------------------
__global__ __launch_bounds__(128)
void attn_kernel(const __bf16* __restrict__ Qg, const __bf16* __restrict__ Kg,
                 const __bf16* __restrict__ Vt, __bf16* __restrict__ attO) {
  __shared__ __align__(16) __bf16 Plds[4][16 * 64];

  const int tid  = threadIdx.x;
  const int w    = tid >> 5;
  const int lane = tid & 31;
  const int l16  = lane & 15;
  const int hi   = lane >> 4;
  const int qt   = blockIdx.x;               // 64-row q tile index
  const int bh   = blockIdx.y;               // b*H + h
  const int b    = bh >> 4;
  const int h    = bh & (H_ - 1);
  const int qbase = qt * 64 + w * 16;

  const __bf16* Qrow = Qg + (size_t)bh * T_ * D_;
  const __bf16* Krow = Kg + (size_t)bh * T_ * D_;
  const __bf16* Vrow = Vt + (size_t)bh * D_ * T_;

  // Q A-fragments (16 rows x 64 d), resident across the kv loop.
  v16bf qf[2];
#pragma unroll
  for (int f = 0; f < 2; ++f) {
    const __bf16* p = Qrow + (size_t)(qbase + l16) * D_ + f * 32 + hi * 8;
    qf[f] = mk16(p, p + 16);
  }

  float mrow[8], lrow[8];
  v8f o[4];
#pragma unroll
  for (int r = 0; r < 8; ++r) { mrow[r] = -INFINITY; lrow[r] = 0.f; }
#pragma unroll
  for (int dt = 0; dt < 4; ++dt) o[dt] = v8f{};

  const int njt = qt + 1;                    // causal: kv tiles 0..qt
  for (int j = 0; j < njt; ++j) {
    const int kb = j * 64;

    // ---- S = Q K^T. B-fragment lane = kv column, contiguous over d. ----
    v8f s[4];
#pragma unroll
    for (int nt = 0; nt < 4; ++nt) {
      const __bf16* p = Krow + (size_t)(kb + nt * 16 + l16) * D_ + hi * 8;
      v16bf k0f = mk16(p, p + 16);           // d 0..31
      v16bf k1f = mk16(p + 32, p + 48);      // d 32..63
      v8f z = v8f{};
      z = wmma_bf16(qf[0], k0f, z);
      z = wmma_bf16(qf[1], k1f, z);
      s[nt] = z;
    }

    // ---- scale + causal mask ----
    const bool diag = (j == qt);
#pragma unroll
    for (int nt = 0; nt < 4; ++nt) {
      const int kvn = kb + nt * 16 + l16;
#pragma unroll
      for (int r = 0; r < 8; ++r) {
        float v = s[nt][r] * 0.125f;         // 1/sqrt(64)
        if (diag && (kvn > qbase + r + hi * 8)) v = -INFINITY;
        s[nt][r] = v;
      }
    }

    // ---- online softmax: per-row max/sum across 16 lanes of the half ----
#pragma unroll
    for (int r = 0; r < 8; ++r) {
      float mx = fmaxf(fmaxf(s[0][r], s[1][r]), fmaxf(s[2][r], s[3][r]));
#pragma unroll
      for (int off = 1; off < 16; off <<= 1)
        mx = fmaxf(mx, __shfl_xor(mx, off, 32));
      const float mnew = fmaxf(mrow[r], mx);
      const float corr = __expf(mrow[r] - mnew);
      mrow[r] = mnew;
      float rs = 0.f;
#pragma unroll
      for (int nt = 0; nt < 4; ++nt) {
        const float pe = __expf(s[nt][r] - mnew);
        s[nt][r] = pe;
        rs += pe;
      }
#pragma unroll
      for (int off = 1; off < 16; off <<= 1)
        rs += __shfl_xor(rs, off, 32);
      lrow[r] = lrow[r] * corr + rs;
#pragma unroll
      for (int dt = 0; dt < 4; ++dt) o[dt][r] *= corr;
    }

    // ---- P: C-layout f32 -> LDS bf16 [q row][kv] (A-layout source) ----
#pragma unroll
    for (int nt = 0; nt < 4; ++nt)
#pragma unroll
      for (int r = 0; r < 8; ++r)
        Plds[w][(r + hi * 8) * 64 + nt * 16 + l16] = (__bf16)s[nt][r];
    __syncthreads();

    v16bf pf[2];
#pragma unroll
    for (int f = 0; f < 2; ++f) {
      const __bf16* p = &Plds[w][l16 * 64 + f * 32 + hi * 8];
      pf[f] = mk16(p, p + 16);
    }

    // ---- O += P V.  V stored [d][t]: lane = d column, contiguous over kv. --
#pragma unroll
    for (int dt = 0; dt < 4; ++dt) {
      const __bf16* vp = Vrow + (size_t)(dt * 16 + l16) * T_ + kb + hi * 8;
      v16bf v0f = mk16(vp, vp + 16);         // kv 0..31
      v16bf v1f = mk16(vp + 32, vp + 48);    // kv 32..63
      o[dt] = wmma_bf16(pf[0], v0f, o[dt]);
      o[dt] = wmma_bf16(pf[1], v1f, o[dt]);
    }
    __syncthreads();
  }

  // ---- normalize + write bf16 [B, T, C] (head-interleaved) ----
  float invl[8];
#pragma unroll
  for (int r = 0; r < 8; ++r) invl[r] = 1.f / lrow[r];
#pragma unroll
  for (int dt = 0; dt < 4; ++dt)
#pragma unroll
    for (int r = 0; r < 8; ++r) {
      const int t   = qbase + r + hi * 8;
      const int col = h * D_ + dt * 16 + l16;
      attO[((size_t)b * T_ + t) * C_ + col] = (__bf16)(o[dt][r] * invl[r]);
    }
}

// ---------------------------------------------------------------------------
extern "C" void kernel_launch(void* const* d_in, const int* in_sizes, int n_in,
                              void* d_out, int out_size, void* d_ws,
                              size_t ws_size, hipStream_t stream) {
  const float* x  = (const float*)d_in[0];
  const float* wq = (const float*)d_in[1];
  const float* bq = (const float*)d_in[2];
  const float* wk = (const float*)d_in[3];
  const float* bk = (const float*)d_in[4];
  const float* wv = (const float*)d_in[5];
  const float* bv = (const float*)d_in[6];
  const float* wp = (const float*)d_in[7];
  const float* bp = (const float*)d_in[8];
  float* out = (float*)d_out;

  // Workspace layout (requires >= 48 MB):
  char* ws = (char*)d_ws;
  const size_t WSZ = (size_t)C_ * C_ * 2;          // 2 MB per weight
  __bf16* wqT = (__bf16*)(ws + 0 * WSZ);
  __bf16* wkT = (__bf16*)(ws + 1 * WSZ);
  __bf16* wvT = (__bf16*)(ws + 2 * WSZ);
  __bf16* wpT = (__bf16*)(ws + 3 * WSZ);
  size_t off = 4 * WSZ;
  const size_t BUF = (size_t)B_ * T_ * C_ * 2;     // 8 MB each
  __bf16* Xb  = (__bf16*)(ws + off); off += BUF;
  __bf16* Qg  = (__bf16*)(ws + off); off += BUF;
  __bf16* Kg  = (__bf16*)(ws + off); off += BUF;
  __bf16* Vt  = (__bf16*)(ws + off); off += BUF;
  __bf16* aOb = (__bf16*)(ws + off);               // 8 MB

  const dim3 cgrid(C_ * C_ / 256);
  wt_bf16_kernel<<<cgrid, 256, 0, stream>>>(wq, wqT);
  wt_bf16_kernel<<<cgrid, 256, 0, stream>>>(wk, wkT);
  wt_bf16_kernel<<<cgrid, 256, 0, stream>>>(wv, wvT);
  wt_bf16_kernel<<<cgrid, 256, 0, stream>>>(wp, wpT);
  x_bf16_kernel<<<dim3(M_ * C_ / 4 / 256), 256, 0, stream>>>(x, Xb);

  const dim3 ggrid(M_ / 128, C_ / 128);
  gemm_kernel<<<ggrid, 256, 0, stream>>>(Xb, wqT, bq, nullptr, Qg, 1);
  gemm_kernel<<<ggrid, 256, 0, stream>>>(Xb, wkT, bk, nullptr, Kg, 1);
  gemm_kernel<<<ggrid, 256, 0, stream>>>(Xb, wvT, bv, nullptr, Vt, 2);

  const dim3 agrid(T_ / 64, B_ * H_);
  attn_kernel<<<agrid, 128, 0, stream>>>(Qg, Kg, Vt, aOb);

  gemm_kernel<<<ggrid, 256, 0, stream>>>(aOb, wpT, bp, out, nullptr, 0);
}
